// GNN_60988535603354
// MI455X (gfx1250) — matrix operand order repaired and checked by reading, not compiled, not run on hardware
//
#include <hip/hip_runtime.h>

typedef __attribute__((ext_vector_type(16))) __bf16 v16bf;
typedef __attribute__((ext_vector_type(8)))  __bf16 v8bf;
typedef __attribute__((ext_vector_type(2)))  __bf16 v2bf;
typedef __attribute__((ext_vector_type(8)))  float  v8f;
typedef unsigned short u16;
typedef unsigned int   u32;

#define DEVINL __device__ __forceinline__

#if __has_builtin(__builtin_amdgcn_cvt_pk_bf16_f32)
#define HAVE_CVT_PK_BF16 1
DEVINL v2bf cvt2bf(float a, float b) {            // v_cvt_pk_bf16_f32 (RNE)
  return __builtin_amdgcn_cvt_pk_bf16_f32(a, b);
}
DEVINL u16 f2bf(float f) {
  union { v2bf v; u16 s[2]; } u; u.v = cvt2bf(f, f);
  return u.s[0];
}
#else
#define HAVE_CVT_PK_BF16 0
DEVINL u16 f2bf(float f) {
  union { float f; u32 u; } v; v.f = f;
  u32 r = v.u + 0x7FFFu + ((v.u >> 16) & 1u);     // round-to-nearest-even
  return (u16)(r >> 16);
}
#endif

DEVINL float bf2f(u16 h) {
  union { u32 u; float f; } v; v.u = ((u32)h) << 16;
  return v.f;
}
DEVINL v16bf cat8(v8bf lo, v8bf hi) {
  return __builtin_shufflevector(lo, hi, 0,1,2,3,4,5,6,7,8,9,10,11,12,13,14,15);
}

// ---------------------------------------------------------------------------
// Fused GEMM:  Y = relu( [Af32 | Abf] @ WT^T + bias ), f32 accumulation via
// v_wmma_f32_16x16x32_bf16.  WT is bf16 [Ntot][Ktot] (output-column major) so
// B fragments are contiguous b128 loads.  Af32 is the fp32 edge-aggregation
// half (k < Ksplit, converted with v_cvt_pk_bf16_f32 in-register); Abf is the
// bf16 feature half (k >= Ksplit).  Block = 8 waves = 32 rows x 256 cols;
// each wave: 1 A-fragment reused across 4 WMMAs (4 N-tiles).
// ---------------------------------------------------------------------------
__global__ __launch_bounds__(256, 2)
void gemm_bf16_wmma(const float* __restrict__ Af32, const u16* __restrict__ Abf,
                    const u16* __restrict__ WT, const float* __restrict__ bias,
                    u16* __restrict__ Ybf,
                    int M, int Ntot, int Ktot, int Ksplit, int relu) {
  (void)M;
  const int lane = threadIdx.x & 31;
  const int wave = threadIdx.x >> 5;
  const int hf   = lane >> 4;          // half-wave select
  const int l16  = lane & 15;
  const int m_base = blockIdx.x * 32 + (wave >> 2) * 16;
  const int n_base = blockIdx.y * 256 + (wave & 3) * 64;
  const int Kx  = Ktot - Ksplit;
  const int row = m_base + l16;

  v8f acc[4] = {};

  for (int k0 = 0; k0 < Ktot; k0 += 32) {
    // ---- A fragment: 16-bit A 16x32 layout (VGPR0-3: K=hf*8+0..7,
    //      VGPR4-7: K=16+hf*8+0..7), row = lane%16 ----
    v16bf a;
    if (k0 < Ksplit) {
      const float* ap = Af32 + (size_t)row * Ksplit + k0 + hf * 8;
#if HAVE_CVT_PK_BF16
      union { v8bf v; v2bf p[4]; } lo, hi;
#pragma unroll
      for (int j = 0; j < 4; ++j) {
        lo.p[j] = cvt2bf(ap[2 * j],      ap[2 * j + 1]);
        hi.p[j] = cvt2bf(ap[16 + 2 * j], ap[17 + 2 * j]);
      }
      a = cat8(lo.v, hi.v);
#else
      union { v8bf v; u16 s[8]; } lo, hi;
#pragma unroll
      for (int j = 0; j < 8; ++j) {
        lo.s[j] = f2bf(ap[j]);
        hi.s[j] = f2bf(ap[16 + j]);
      }
      a = cat8(lo.v, hi.v);
#endif
    } else {
      const u16* ap = Abf + (size_t)row * Kx + (k0 - Ksplit) + hf * 8;
      a = cat8(*(const v8bf*)ap, *(const v8bf*)(ap + 16));
    }
    if (k0 + 32 < Ktot && k0 + 32 >= Ksplit)
      __builtin_prefetch(Abf + (size_t)row * Kx + (k0 + 32 - Ksplit), 0, 1);

    // ---- B fragments + WMMA: B 32x16 layout (lanes = N, K = hf*8+0..7, 16+hf*8+0..7)
#pragma unroll
    for (int t = 0; t < 4; ++t) {
      const u16* bp = WT + (size_t)(n_base + t * 16 + l16) * Ktot + k0 + hf * 8;
      v16bf b = cat8(*(const v8bf*)bp, *(const v8bf*)(bp + 16));
      acc[t] = __builtin_amdgcn_wmma_f32_16x16x32_bf16(
          /*neg_a=*/false, a, /*neg_b=*/false, b,
          /*c_mod=*/(short)0, acc[t], /*reuse_a=*/false, /*reuse_b=*/false);
    }
  }

  // ---- Epilogue: D layout row = vgpr + 8*half, col = n_tile + lane%16 ----
#pragma unroll
  for (int t = 0; t < 4; ++t) {
    const int col = n_base + t * 16 + l16;
    const float bv = bias[col];
#pragma unroll
    for (int i = 0; i < 8; ++i) {
      const int r = m_base + hf * 8 + i;
      float v = acc[t][i] + bv;
      if (relu) v = v > 0.f ? v : 0.f;
      Ybf[(size_t)r * Ntot + col] = f2bf(v);
    }
  }
}

// ---------------------------------------------------------------------------
// Edge aggregation: agg[dst] += x[src]  (x bf16, agg fp32, hardware f32 atomics)
// One thread per (edge, 8-channel chunk); 16B vector gather per thread.
// ---------------------------------------------------------------------------
__global__ void edge_agg_kernel(const int* __restrict__ src, const int* __restrict__ dst,
                                const u16* __restrict__ x, float* __restrict__ agg,
                                int C, int chunk_shift, long long total) {
  long long tid = (long long)blockIdx.x * blockDim.x + threadIdx.x;
  if (tid >= total) return;
  const int e  = (int)(tid >> chunk_shift);
  const int c0 = ((int)tid & ((1 << chunk_shift) - 1)) << 3;
  const int s = src[e], d = dst[e];
  union { v8bf v; u16 sv[8]; } u;
  u.v = *(const v8bf*)(x + (size_t)s * C + c0);
  float* ap = agg + (size_t)d * C + c0;
#pragma unroll
  for (int j = 0; j < 8; ++j) atomicAdd(ap + j, bf2f(u.sv[j]));
}

// global_add_pool: g[batch[n]] += x[n]
__global__ void pool_kernel(const int* __restrict__ batch, const u16* __restrict__ x,
                            float* __restrict__ g, int C, int chunk_shift,
                            long long total) {
  long long tid = (long long)blockIdx.x * blockDim.x + threadIdx.x;
  if (tid >= total) return;
  const int n  = (int)(tid >> chunk_shift);
  const int c0 = ((int)tid & ((1 << chunk_shift) - 1)) << 3;
  const int b = batch[n];
  union { v8bf v; u16 sv[8]; } u;
  u.v = *(const v8bf*)(x + (size_t)n * C + c0);
  float* gp = g + (size_t)b * C + c0;
#pragma unroll
  for (int j = 0; j < 8; ++j) atomicAdd(gp + j, bf2f(u.sv[j]));
}

// x0 = bf16(embed[x_idx])
__global__ void embed_kernel(const int* __restrict__ idx, const float* __restrict__ tab,
                             u16* __restrict__ xb, int n_nodes, int emb) {
  int tid = blockIdx.x * blockDim.x + threadIdx.x;
  if (tid >= n_nodes * emb) return;
  int n = tid / emb, c = tid - n * emb;
  xb[tid] = f2bf(tab[(size_t)idx[n] * emb + c]);
}

// WT[n][k] = bf16( k < K ? w_rel[k][n] : w_root[k-K][n] )   (fused-K transpose)
__global__ void prep_wcatT_kernel(const float* __restrict__ wrel,
                                  const float* __restrict__ wroot,
                                  u16* __restrict__ WT, int K, int N) {
  int tid = blockIdx.x * blockDim.x + threadIdx.x;
  int K2 = 2 * K;
  if (tid >= N * K2) return;
  int n = tid / K2, k = tid - n * K2;
  float v = (k < K) ? wrel[(size_t)k * N + n] : wroot[(size_t)(k - K) * N + n];
  WT[(size_t)n * K2 + k] = f2bf(v);
}

// WT[n][k] = bf16(w[k][n])
__global__ void prep_wT_kernel(const float* __restrict__ w, u16* __restrict__ WT,
                               int K, int N) {
  int tid = blockIdx.x * blockDim.x + threadIdx.x;
  if (tid >= N * K) return;
  int n = tid / K, k = tid - n * K;
  WT[(size_t)n * K + k] = f2bf(w[(size_t)k * N + n]);
}

__global__ void cvt_kernel(const float* __restrict__ in, u16* __restrict__ out, int n) {
  int tid = blockIdx.x * blockDim.x + threadIdx.x;
  if (tid < n) out[tid] = f2bf(in[tid]);
}

// y[g] = h[g] . w_out + b_out  — one wave32 per graph, shfl_xor butterfly reduce
__global__ void out_kernel(const u16* __restrict__ h, const float* __restrict__ w,
                           const float* __restrict__ b, float* __restrict__ y,
                           int G, int K) {
  const int lane = threadIdx.x & 31;
  const int wv   = threadIdx.x >> 5;
  const int g = blockIdx.x * (blockDim.x >> 5) + wv;
  if (g >= G) return;
  float s = 0.f;
  for (int k = lane; k < K; k += 32) s += bf2f(h[(size_t)g * K + k]) * w[k];
#pragma unroll
  for (int m = 16; m >= 1; m >>= 1) s += __shfl_xor(s, m, 32);
  if (lane == 0) y[g] = s + b[0];
}

extern "C" void kernel_launch(void* const* d_in, const int* in_sizes, int n_in,
                              void* d_out, int out_size, void* d_ws, size_t ws_size,
                              hipStream_t stream) {
  (void)in_sizes; (void)n_in; (void)out_size; (void)ws_size;
  enum { EMB = 64, L_CONV = 5, C = 256, C2 = 512, L_LIN = 2,
         N_NODES = 40000, N_EDGES = 640000, N_GRAPHS = 128 };

  const int*   x_idx  = (const int*)d_in[0];
  const int*   ei     = (const int*)d_in[1];
  const int*   src    = ei;
  const int*   dst    = ei + N_EDGES;
  const int*   batch  = (const int*)d_in[2];
  const float* embed  = (const float*)d_in[3];
  const float* w_rel0 = (const float*)d_in[4];
  const float* b_rel0 = (const float*)d_in[5];
  const float* w_root0= (const float*)d_in[6];
  const float* w_rel  = (const float*)d_in[7];
  const float* b_rel  = (const float*)d_in[8];
  const float* w_root = (const float*)d_in[9];
  const float* w_lin0 = (const float*)d_in[10];
  const float* b_lin0 = (const float*)d_in[11];
  const float* w_lin  = (const float*)d_in[12];
  const float* b_lin  = (const float*)d_in[13];
  const float* w_out  = (const float*)d_in[14];
  const float* b_out  = (const float*)d_in[15];
  float* out = (float*)d_out;

  char* ws = (char*)d_ws;
  size_t off = 0;
  auto alloc = [&](size_t bytes) {
    char* p = ws + off;
    off = (off + bytes + 255) & ~(size_t)255;
    return p;
  };

  u16*   xb0   = (u16*)alloc((size_t)N_NODES * C * 2);
  u16*   xb1   = (u16*)alloc((size_t)N_NODES * C * 2);
  float* agg   = (float*)alloc((size_t)N_NODES * C * 4);
  u16*   x0b   = (u16*)alloc((size_t)N_NODES * EMB * 2);
  u16*   w0T   = (u16*)alloc((size_t)C * 2 * EMB * 2);
  u16*   wT[L_CONV];
  for (int i = 0; i < L_CONV; ++i) wT[i] = (u16*)alloc((size_t)C * 2 * C * 2);
  u16*   lin0T = (u16*)alloc((size_t)C2 * C * 2);
  u16*   linT[L_LIN];
  for (int i = 0; i < L_LIN; ++i) linT[i] = (u16*)alloc((size_t)C2 * C2 * 2);
  float* gf    = (float*)alloc((size_t)N_GRAPHS * C * 4);
  u16*   gb    = (u16*)alloc((size_t)N_GRAPHS * C * 2);
  u16*   h0    = (u16*)alloc((size_t)N_GRAPHS * C2 * 2);
  u16*   h1    = (u16*)alloc((size_t)N_GRAPHS * C2 * 2);

  const int TB = 256;

  // ---- one-time weight prep (bf16, output-column-major, rel/root fused in K)
  prep_wcatT_kernel<<<(C * 2 * EMB + TB - 1) / TB, TB, 0, stream>>>(w_rel0, w_root0, w0T, EMB, C);
  for (int i = 0; i < L_CONV; ++i)
    prep_wcatT_kernel<<<(C * 2 * C + TB - 1) / TB, TB, 0, stream>>>(
        w_rel + (size_t)i * C * C, w_root + (size_t)i * C * C, wT[i], C, C);
  prep_wT_kernel<<<(C2 * C + TB - 1) / TB, TB, 0, stream>>>(w_lin0, lin0T, C, C2);
  for (int i = 0; i < L_LIN; ++i)
    prep_wT_kernel<<<(C2 * C2 + TB - 1) / TB, TB, 0, stream>>>(
        w_lin + (size_t)i * C2 * C2, linT[i], C2, C2);

  // ---- embedding gather
  embed_kernel<<<(N_NODES * EMB + TB - 1) / TB, TB, 0, stream>>>(x_idx, embed, x0b, N_NODES, EMB);

  // ---- in_conv: GraphConv(EMB -> C)
  hipMemsetAsync(agg, 0, (size_t)N_NODES * EMB * 4, stream);
  {
    long long tot = (long long)N_EDGES * (EMB / 8);
    edge_agg_kernel<<<(unsigned)((tot + TB - 1) / TB), TB, 0, stream>>>(src, dst, x0b, agg, EMB, 3, tot);
  }
  gemm_bf16_wmma<<<dim3(N_NODES / 32, 1), TB, 0, stream>>>(
      agg, x0b, w0T, b_rel0, xb0, N_NODES, C, 2 * EMB, EMB, 1);

  // ---- stacked GraphConv(C -> C) x L_CONV
  u16* cur = xb0; u16* nxt = xb1;
  for (int i = 0; i < L_CONV; ++i) {
    hipMemsetAsync(agg, 0, (size_t)N_NODES * C * 4, stream);
    long long tot = (long long)N_EDGES * (C / 8);
    edge_agg_kernel<<<(unsigned)((tot + TB - 1) / TB), TB, 0, stream>>>(src, dst, cur, agg, C, 5, tot);
    gemm_bf16_wmma<<<dim3(N_NODES / 32, 1), TB, 0, stream>>>(
        agg, cur, wT[i], b_rel + (size_t)i * C, nxt, N_NODES, C, 2 * C, C, 1);
    u16* t = cur; cur = nxt; nxt = t;
  }

  // ---- global_add_pool
  hipMemsetAsync(gf, 0, (size_t)N_GRAPHS * C * 4, stream);
  {
    long long tot = (long long)N_NODES * (C / 8);
    pool_kernel<<<(unsigned)((tot + TB - 1) / TB), TB, 0, stream>>>(batch, cur, gf, C, 5, tot);
  }
  cvt_kernel<<<(N_GRAPHS * C + TB - 1) / TB, TB, 0, stream>>>(gf, gb, N_GRAPHS * C);

  // ---- MLP head (WMMA as well: M=128)
  gemm_bf16_wmma<<<dim3(N_GRAPHS / 32, C2 / 256), TB, 0, stream>>>(
      nullptr, gb, lin0T, b_lin0, h0, N_GRAPHS, C2, C, 0, 1);
  u16* hc = h0; u16* hn = h1;
  for (int i = 0; i < L_LIN; ++i) {
    gemm_bf16_wmma<<<dim3(N_GRAPHS / 32, C2 / 256), TB, 0, stream>>>(
        nullptr, hc, linT[i], b_lin + (size_t)i * C2, hn, N_GRAPHS, C2, C2, 0, 1);
    u16* t = hc; hc = hn; hn = t;
  }
  out_kernel<<<dim3((N_GRAPHS + 3) / 4), 128, 0, stream>>>(hc, w_out, b_out, out, N_GRAPHS, C2);
}
